// WindowAttention_76776835383415
// MI455X (gfx1250) — matrix operand order repaired and checked by reading, not compiled, not run on hardware
//
#include <hip/hip_runtime.h>

#define DIM   384
#define NH    12
#define HD    32
#define NTOK  49
#define NPAD  64
#define NWIN  64
#define BW    2048
#define HCOLS 96     // q|k|v columns per head

typedef __bf16 bf16_t;
typedef bf16_t v16bf __attribute__((ext_vector_type(16)));
typedef float  v8f   __attribute__((ext_vector_type(8)));

union FragAB { v16bf v; unsigned short u[16]; uint4 q[2]; };

static __device__ __forceinline__ unsigned short f2bf(float f) {
    union { float f; unsigned u; } x; x.f = f;
    unsigned u = x.u;
    unsigned r = (u + 0x7FFFu + ((u >> 16) & 1u)) >> 16;   // round-to-nearest-even
    return (unsigned short)r;
}

static __device__ __forceinline__ v8f wmma_bf16(const FragAB& a, const FragAB& b, v8f c) {
    return __builtin_amdgcn_wmma_f32_16x16x32_bf16(
        false, a.v, false, b.v, (short)0, c, false, false);
}

// ---------------------------------------------------------------------------
// Kernel 1: qkv_w fp32 [k][n] -> bf16 per-head slices whd[head][c][k],
// c = qsel*32 + d  (B-operand friendly: K contiguous per output column)
// ---------------------------------------------------------------------------
__global__ void wcvt_kernel(const float* __restrict__ w, unsigned short* __restrict__ whd) {
    int idx = blockIdx.x * blockDim.x + threadIdx.x;
    if (idx >= NH * HCOLS * DIM) return;
    int head = idx / (HCOLS * DIM);
    int rem  = idx - head * (HCOLS * DIM);
    int c    = rem / DIM;
    int k    = rem - c * DIM;
    int qsel = c >> 5, d = c & 31;
    int n    = qsel * DIM + head * HD + d;
    whd[idx] = f2bf(w[k * (3 * DIM) + n]);
}

// ---------------------------------------------------------------------------
// Fused kernel: one block = (head, window), 192 threads = 6 waves.
//   Phase 1: qkv head-slice = x(64pad x 384) @ Whd(384 x 96) + b  -> LDS
//   Phase 2: S = scale*q k^T + bias + mask ; softmax ; O = P V -> out
// ---------------------------------------------------------------------------
__global__ void __launch_bounds__(192)
fused_attn_kernel(const float* __restrict__ x, const float* __restrict__ qkv_b,
                  const unsigned short* __restrict__ whd,
                  const float* __restrict__ mask,
                  const float* __restrict__ bias_table,
                  const int* __restrict__ rel_index,
                  float* __restrict__ out) {
    // region A (48KB): phase1 = bf16 x-tile ; phase2 = bm(16K) | sbuf(16K) | ps(8K)
    __shared__ __align__(16) unsigned char smemA[NPAD * DIM * 2];
    __shared__ __align__(16) unsigned short qs[NPAD * HD];   // q [m][d]   4KB
    __shared__ __align__(16) unsigned short ks[NPAD * HD];   // k [m][d]   4KB
    __shared__ __align__(16) unsigned short vt[HD * NPAD];   // v^T [d][m] 4KB

    unsigned short* xs   = (unsigned short*)smemA;
    float*          bm   = (float*)smemA;                       // 16KB
    float*          sbuf = (float*)(smemA + 16384);             // 16KB
    unsigned short* ps   = (unsigned short*)(smemA + 32768);    //  8KB

    const int tid  = threadIdx.x;
    const int head = blockIdx.x;   // fastest dim: 12 head-blocks share L2-hot x window
    const int win  = blockIdx.y;

    // ---- phase 0: stage x window as bf16 with zero row padding -------------
    for (int i = tid; i < (NPAD - NTOK) * DIM; i += 192)
        xs[NTOK * DIM + i] = 0;
    const float4* xg = (const float4*)(x + (size_t)win * NTOK * DIM);
    for (int i = tid; i < NTOK * DIM / 4; i += 192) {
        float4 f = xg[i];
        int o = i * 4;
        xs[o + 0] = f2bf(f.x); xs[o + 1] = f2bf(f.y);
        xs[o + 2] = f2bf(f.z); xs[o + 3] = f2bf(f.w);
    }
    __syncthreads();

    const int wave = tid >> 5;
    const int lane = tid & 31;
    const int l16  = lane & 15;
    const int hi   = (lane >= 16);

    // ---- phase 1: per-head QKV GEMM (wave w owns n-tile w: 16 of 96 cols) --
    {
        const int c = wave * 16 + l16;            // 0..95 ; qsel uniform per wave
        v8f acc[4] = {};
        const unsigned short* wcol = whd + ((size_t)head * HCOLS + c) * DIM;
        for (int kb = 0; kb < DIM; kb += 32) {
            FragAB b;   // B 32x16: lane = fixed N, dual 8-elem K chunks
            const unsigned short* br = wcol + kb + (hi ? 8 : 0);
            b.q[0] = *(const uint4*)(br);
            b.q[1] = *(const uint4*)(br + 16);
            for (int mt = 0; mt < 4; ++mt) {
                FragAB a;   // A 16x32: row = l16, dual 8-elem K chunks
                const unsigned short* ar = xs + (mt * 16 + l16) * DIM + kb + (hi ? 8 : 0);
                a.q[0] = *(const uint4*)(ar);
                a.q[1] = *(const uint4*)(ar + 16);
                acc[mt] = wmma_bf16(a, b, acc[mt]);
            }
        }
        const int qsel = c >> 5, d = c & 31;
        const float bias = qkv_b[qsel * DIM + head * HD + d];
        for (int mt = 0; mt < 4; ++mt)
            for (int j = 0; j < 8; ++j) {
                int row = mt * 16 + j + (hi ? 8 : 0);
                unsigned short v = f2bf(acc[mt][j] + bias);
                if (qsel == 0)      qs[row * HD + d] = v;
                else if (qsel == 1) ks[row * HD + d] = v;
                else                vt[d * NPAD + row] = v;
            }
    }
    __syncthreads();   // x-tile dead; region A reused below

    // ---- phase 2a: combined mask + relative-position bias ------------------
    const float* mwin = mask + (size_t)(win & (NWIN - 1)) * NTOK * NTOK;
    for (int i = tid; i < NPAD * NPAD; i += 192) {
        int r = i >> 6, cc = i & 63;
        float v;
        if (cc >= NTOK)      v = -1e30f;   // padded keys: killed by softmax
        else if (r >= NTOK)  v = 0.0f;     // padded query rows: discarded at store
        else v = mwin[r * NTOK + cc] + bias_table[rel_index[r * NTOK + cc] * NH + head];
        bm[i] = v;
    }
    __syncthreads();

    // ---- phase 2b: attention, waves 0..3 own one 16-row strip each ---------
    if (wave < 4) {
        const int mi = wave;
        const float scale = 0.17677669529663687f;   // 1/sqrt(32)
        float* srow = sbuf + wave * (16 * NPAD);

        // S = scale * q @ k^T + bm   (K = 32 -> single wmma step)
        for (int nt = 0; nt < 4; ++nt) {
            FragAB a, b;
            const unsigned short* qr = qs + (mi * 16 + l16) * HD + (hi ? 8 : 0);
            a.q[0] = *(const uint4*)(qr);
            a.q[1] = *(const uint4*)(qr + 16);
            const unsigned short* kr = ks + (nt * 16 + l16) * HD + (hi ? 8 : 0);
            b.q[0] = *(const uint4*)(kr);
            b.q[1] = *(const uint4*)(kr + 16);
            v8f c = {};
            c = wmma_bf16(a, b, c);
            for (int j = 0; j < 8; ++j) {
                int r = j + (hi ? 8 : 0);
                int gcol = nt * 16 + l16;
                srow[r * NPAD + gcol] = c[j] * scale + bm[(mi * 16 + r) * NPAD + gcol];
            }
        }
        asm volatile("s_wait_dscnt 0" ::: "memory");

        // row softmax (lanes 0..15 own one row each), write P as bf16
        if (lane < 16) {
            float* row = srow + lane * NPAD;
            float mx = -1e30f;
            for (int c = 0; c < NPAD; ++c) mx = fmaxf(mx, row[c]);
            float sum = 0.0f;
            for (int c = 0; c < NPAD; ++c) { float e = __expf(row[c] - mx); row[c] = e; sum += e; }
            float inv = 1.0f / sum;
            unsigned short* prow = ps + (mi * 16 + lane) * NPAD;
            for (int c = 0; c < NPAD; ++c) prow[c] = f2bf(row[c] * inv);
        }
        asm volatile("s_wait_dscnt 0" ::: "memory");

        // O = P @ V   (K = 64 -> two k-steps, N = 32 -> two n-tiles)
        v8f o[2] = {};
        for (int kt = 0; kt < 2; ++kt) {
            FragAB a;
            const unsigned short* pr = ps + (mi * 16 + l16) * NPAD + kt * 32 + (hi ? 8 : 0);
            a.q[0] = *(const uint4*)(pr);
            a.q[1] = *(const uint4*)(pr + 16);
            for (int nt = 0; nt < 2; ++nt) {
                FragAB b;
                const unsigned short* vr = vt + (nt * 16 + l16) * NPAD + kt * 32 + (hi ? 8 : 0);
                b.q[0] = *(const uint4*)(vr);
                b.q[1] = *(const uint4*)(vr + 16);
                o[nt] = wmma_bf16(a, b, o[nt]);
            }
        }

        // store O in (Bw, n, c) layout, c = head*32 + d
        float* obase = out + (size_t)win * NTOK * DIM + head * HD;
        for (int nt = 0; nt < 2; ++nt)
            for (int j = 0; j < 8; ++j) {
                int r = mi * 16 + j + (hi ? 8 : 0);
                if (r < NTOK) obase[(size_t)r * DIM + nt * 16 + l16] = o[nt][j];
            }
    }
}

// ---------------------------------------------------------------------------
extern "C" void kernel_launch(void* const* d_in, const int* in_sizes, int n_in,
                              void* d_out, int out_size, void* d_ws, size_t ws_size,
                              hipStream_t stream) {
    (void)in_sizes; (void)n_in; (void)out_size; (void)ws_size;
    const float* x        = (const float*)d_in[0];
    const float* mask     = (const float*)d_in[1];
    const float* qkv_w    = (const float*)d_in[2];
    const float* qkv_b    = (const float*)d_in[3];
    const float* bias_tab = (const float*)d_in[4];
    const int*   rel_idx  = (const int*)d_in[5];
    float* out = (float*)d_out;

    // scratch: 884736 B of bf16 per-head-sliced W (L2-resident, reused 2048x)
    unsigned short* whd = (unsigned short*)d_ws;

    int total = NH * HCOLS * DIM;
    wcvt_kernel<<<(total + 255) / 256, 256, 0, stream>>>(qkv_w, whd);
    fused_attn_kernel<<<dim3(NH, BW), 192, 0, stream>>>(
        x, qkv_b, whd, mask, bias_tab, rel_idx, out);
}
